// InnerResShiftTriple_6871947674014
// MI455X (gfx1250) — matrix operand order repaired and checked by reading, not compiled, not run on hardware
//
#include <hip/hip_runtime.h>
#include <hip/hip_bf16.h>
#include <math.h>

typedef _Float16 half_t;
typedef __attribute__((ext_vector_type(16))) _Float16 v16h;
typedef __attribute__((ext_vector_type(8)))  _Float16 v8h;
typedef __attribute__((ext_vector_type(8)))  float    v8f;

#define B_     8
#define C_     256
#define C2_    512
#define HW_    64
#define L_     4096
#define NEG_   (-1.0e9f)

// ---------------------------------------------------------------------------
// Workspace layout (bytes). Aliased regions noted.
//   idx      : B*L*4                  =    131,072
//   W1h      : 256*512*9*2            =  2,359,296
//   W2h      : 256*256*9*2            =  1,179,648
//   latn/h   : B*C*L*2                = 16,777,216   (latn dead after sim -> reuse as h)
//   Fcom/c2o : B*2C*L*2 == B*C*L*4    = 33,554,432   (Fcom dead after conv1 -> conv2 out)
//   conv1out : B*C*L*4                = 33,554,432
// total ~87.6 MB
// ---------------------------------------------------------------------------
#define OFF_IDX   ((size_t)0)
#define OFF_W1H   ((size_t)131072)
#define OFF_W2H   ((size_t)(OFF_W1H + 2359296))
#define OFF_LATN  ((size_t)(OFF_W2H + 1179648))
#define OFF_FCOM  ((size_t)(OFF_LATN + 16777216))
#define OFF_C1O   ((size_t)(OFF_FCOM + 33554432))

// ---------------------------------------------------------------------------
__global__ void cvt_f16_kernel(const float* __restrict__ w, half_t* __restrict__ wh, int n) {
    int t = blockIdx.x * 256 + threadIdx.x;
    if (t < n) wh[t] = (half_t)w[t];
}

// latn[b,c,l] = F_s[b,c,l] / (||F_s[b,:,l]|| + 1e-8), stored f16
__global__ void normalize_kernel(const float* __restrict__ inp, half_t* __restrict__ latn) {
    int t = blockIdx.x * 256 + threadIdx.x;          // over B*L
    int b = t >> 12;
    int l = t & (L_ - 1);
    const float* fs = inp + ((size_t)b * C2_ + C_) * L_ + l;
    float ss = 0.0f;
    for (int c = 0; c < C_; ++c) { float x = fs[(size_t)c * L_]; ss += x * x; }
    float inv = 1.0f / (sqrtf(ss) + 1.0e-8f);
    half_t* o = latn + (size_t)b * C_ * L_ + l;
    for (int c = 0; c < C_; ++c) o[(size_t)c * L_] = (half_t)(fs[(size_t)c * L_] * inv);
}

// ---------------------------------------------------------------------------
// sim = latn^T latn per batch, masked argmax over columns, fused (sim never in HBM).
// Block: 256 threads / 8 waves, owns 128 rows. Per 32-col tile each wave runs
// 2 independent accumulators (2x16x16) over K=256 -> 16 WMMAs between barriers.
// Register-staged pipeline: global loads for tile n+1 issued before computing tile n.
__global__ void sim_argmax_kernel(const half_t* __restrict__ latn,
                                  const int* __restrict__ mask,
                                  int* __restrict__ idxOut) {
    int bid = blockIdx.x;
    int mt = bid & 31;
    int b  = bid >> 5;
    int m0 = mt * 128;
    int tid  = threadIdx.x;
    int wave = tid >> 5, lane = tid & 31;
    int g = lane >> 4, ln = lane & 15;

    __shared__ __align__(16) half_t As[128 * 256];   // [m][k] 64 KB
    __shared__ __align__(16) half_t Bs[32 * 256];    // [n][k] 16 KB

    const half_t* lb = latn + (size_t)b * C_ * L_;

    // Load A panel (rows m0..m0+127, all K). v8h global loads along m, scalar LDS scatter.
    {
        int mseg = tid & 15;                 // 8-half segment along m
        int kb   = tid >> 4;                 // 0..15
        for (int j = 0; j < 16; ++j) {
            int k = kb + 16 * j;
            v8h v = *(const v8h*)(lb + (size_t)k * L_ + m0 + mseg * 8);
            for (int i = 0; i < 8; ++i) As[(mseg * 8 + i) * 256 + k] = v[i];
        }
    }

    // B-tile staging: 32 cols x 256 k = 8192 halves; 4 v8h per thread.
    const int bseg = tid & 3;                // 8-half segment along n
    const int bkb  = tid >> 2;               // 0..63
    v8h breg[4];
    auto loadB = [&](int n0) {
        for (int j = 0; j < 4; ++j)
            breg[j] = *(const v8h*)(lb + (size_t)(bkb + 64 * j) * L_ + n0 + bseg * 8);
    };
    auto storeB = [&]() {
        for (int j = 0; j < 4; ++j) {
            int k = bkb + 64 * j;
            for (int i = 0; i < 8; ++i) Bs[(bseg * 8 + i) * 256 + k] = breg[j][i];
        }
    };

    loadB(0);
    storeB();
    __syncthreads();

    float best[8];
    int   bidx[8];
    for (int r = 0; r < 8; ++r) { best[r] = -3.0e38f; bidx[r] = 0; }

    const half_t* arow  = &As[(wave * 16 + ln) * 256];
    const half_t* brow0 = &Bs[ln * 256];
    const half_t* brow1 = &Bs[(16 + ln) * 256];

    for (int n0 = 0; n0 < L_; n0 += 32) {
        bool more = (n0 + 32) < L_;
        if (more) loadB(n0 + 32);            // global loads overlap the WMMA phase

        v8f acc0 = {}, acc1 = {};
        for (int k0 = 0; k0 < 256; k0 += 32) {
            v8h alo = *(const v8h*)(arow + k0 + 8 * g);
            v8h ahi = *(const v8h*)(arow + k0 + 16 + 8 * g);
            v8h b0l = *(const v8h*)(brow0 + k0 + 8 * g);
            v8h b0h = *(const v8h*)(brow0 + k0 + 16 + 8 * g);
            v8h b1l = *(const v8h*)(brow1 + k0 + 8 * g);
            v8h b1h = *(const v8h*)(brow1 + k0 + 16 + 8 * g);
            v16h a, bf0, bf1;
            for (int i = 0; i < 8; ++i) {
                a[i] = alo[i];  a[i + 8] = ahi[i];
                bf0[i] = b0l[i]; bf0[i + 8] = b0h[i];
                bf1[i] = b1l[i]; bf1[i + 8] = b1h[i];
            }
            acc0 = __builtin_amdgcn_wmma_f32_16x16x32_f16(
                false, a, false, bf0, (short)0, acc0, false, false);
            acc1 = __builtin_amdgcn_wmma_f32_16x16x32_f16(
                false, a, false, bf1, (short)0, acc1, false, false);
        }

        // masked running argmax; n increases -> strict > keeps first occurrence
        int  na = n0 + ln;
        int  nb = na + 16;
        bool ma = mask[na] > 0;
        bool mb = mask[nb] > 0;
        for (int r = 0; r < 8; ++r) {
            float va = ma ? NEG_ : acc0[r];
            if (va > best[r]) { best[r] = va; bidx[r] = na; }
            float vb = mb ? NEG_ : acc1[r];
            if (vb > best[r]) { best[r] = vb; bidx[r] = nb; }
        }

        __syncthreads();
        if (more) storeB();
        __syncthreads();
    }

    // width-16 argmax butterfly (xor masks < 16 never cross the half-wave boundary)
    for (int off = 1; off < 16; off <<= 1) {
        for (int r = 0; r < 8; ++r) {
            float ov = __shfl_xor(best[r], off);
            int   oi = __shfl_xor(bidx[r], off);
            if (ov > best[r] || (ov == best[r] && oi < bidx[r])) { best[r] = ov; bidx[r] = oi; }
        }
    }
    if (ln == 0) {
        int mBase = m0 + wave * 16 + 8 * g;
        for (int r = 0; r < 8; ++r) idxOut[(size_t)b * L_ + mBase + r] = bidx[r];
    }
}

// ---------------------------------------------------------------------------
// F_com = [F_c ; F_c * shift] as f16, shift gathered via idx where mask>0
__global__ void fuse_kernel(const float* __restrict__ inp, const int* __restrict__ mask,
                            const int* __restrict__ idx, half_t* __restrict__ Fcom) {
    size_t t = (size_t)blockIdx.x * 256 + threadIdx.x;   // over B*C*L
    int l = (int)(t & (L_ - 1));
    size_t bc = t >> 12;
    int c = (int)(bc & (C_ - 1));
    int b = (int)(bc >> 8);
    const float* fc = inp + ((size_t)b * C2_ + c) * L_;
    float v = fc[l];
    half_t* ob = Fcom + (size_t)b * C2_ * L_;
    ob[(size_t)c * L_ + l] = (half_t)v;
    float sh = 0.0f;
    if (mask[l] > 0) sh = fc[idx[(size_t)b * L_ + l]];
    ob[(size_t)(C_ + c) * L_ + l] = (half_t)(v * sh);
}

// ---------------------------------------------------------------------------
// Implicit-GEMM 3x3 SAME conv, NCHW.  M=256 (Cout), N=L per batch, K=Cin*9.
// Block: 256 threads / 8 waves, tile M64 x N64; each wave owns 2 independent
// 16x16 output tiles (shared A fragment). Register-staged im2col pipeline.
__global__ void conv_wmma_kernel(const half_t* __restrict__ Xh,   // (B,Cin,L) f16
                                 const half_t* __restrict__ Wh,   // (256,Cin*9) f16
                                 const float* __restrict__ bias,  // (256)
                                 float* __restrict__ Y,           // (B,256,L)
                                 int Cin) {
    const int K = Cin * 9;
    int bid = blockIdx.x;
    int nt = bid & 63;   bid >>= 6;
    int mt = bid & 3;    bid >>= 2;
    int b  = bid;
    int n0 = nt * 64;
    int m0 = mt * 64;

    int tid  = threadIdx.x;
    int wave = tid >> 5, lane = tid & 31;
    int msub = wave & 3, npair = wave >> 2;
    int g = lane >> 4, ln = lane & 15;
    int mBase = m0 + msub * 16;

    __shared__ __align__(16) half_t Bs[64 * 32];   // [n][k] im2col tile (4 KB)

    const half_t* xb   = Xh + (size_t)b * Cin * L_;
    const half_t* wrow = Wh + (size_t)(mBase + ln) * K;

    // per-thread im2col staging coords (constant across K-steps)
    const int nn    = tid & 63;
    const int kbase = tid >> 6;          // 0..3 ; covers kk = kbase + 4*j
    const int l     = n0 + nn;
    const int hb    = l >> 6, wb = l & (HW_ - 1);

    half_t regs[8];
    auto loadRegs = [&](int k0) {
        for (int j = 0; j < 8; ++j) {
            int k  = k0 + kbase + 4 * j;
            int ci = k / 9;
            int r  = k - ci * 9;
            int dy = r / 3 - 1, dx = r % 3 - 1;
            int hh = hb + dy, ww = wb + dx;
            half_t v = (half_t)0.0f;
            if (hh >= 0 && hh < HW_ && ww >= 0 && ww < HW_)
                v = xb[(size_t)ci * L_ + hh * HW_ + ww];
            regs[j] = v;
        }
    };
    auto storeRegs = [&]() {
        for (int j = 0; j < 8; ++j) Bs[nn * 32 + kbase + 4 * j] = regs[j];
    };

    loadRegs(0);
    storeRegs();
    __syncthreads();

    const half_t* brow0 = &Bs[(npair * 32 + ln) * 32];
    const half_t* brow1 = brow0 + 16 * 32;

    v8f acc0 = {}, acc1 = {};
    for (int k0 = 0; k0 < K; k0 += 32) {
        bool more = (k0 + 32) < K;
        if (more) loadRegs(k0 + 32);               // global gathers overlap compute
        __builtin_prefetch((const void*)(wrow + k0 + 256), 0, 1);

        v8h alo = *(const v8h*)(wrow + k0 + 8 * g);
        v8h ahi = *(const v8h*)(wrow + k0 + 16 + 8 * g);
        v8h b0l = *(const v8h*)(brow0 + 8 * g);
        v8h b0h = *(const v8h*)(brow0 + 16 + 8 * g);
        v8h b1l = *(const v8h*)(brow1 + 8 * g);
        v8h b1h = *(const v8h*)(brow1 + 16 + 8 * g);
        v16h a, bf0, bf1;
        for (int i = 0; i < 8; ++i) {
            a[i] = alo[i];  a[i + 8] = ahi[i];
            bf0[i] = b0l[i]; bf0[i + 8] = b0h[i];
            bf1[i] = b1l[i]; bf1[i + 8] = b1h[i];
        }
        acc0 = __builtin_amdgcn_wmma_f32_16x16x32_f16(
            false, a, false, bf0, (short)0, acc0, false, false);
        acc1 = __builtin_amdgcn_wmma_f32_16x16x32_f16(
            false, a, false, bf1, (short)0, acc1, false, false);

        __syncthreads();
        if (more) storeRegs();
        __syncthreads();
    }

    for (int r = 0; r < 8; ++r) {
        int m  = mBase + 8 * g + r;
        int l0 = n0 + npair * 32 + ln;
        float bv = bias[m];
        Y[((size_t)b * C_ + m) * L_ + l0]      = acc0[r] + bv;
        Y[((size_t)b * C_ + m) * L_ + l0 + 16] = acc1[r] + bv;
    }
}

// ---------------------------------------------------------------------------
// Instance norm (+ReLU) -> f16, block per (b,c)
__global__ void inorm_relu_kernel(const float* __restrict__ X, half_t* __restrict__ Yh) {
    int bc = blockIdx.x;
    const float* x = X + (size_t)bc * L_;
    __shared__ float s1[256], s2[256];
    float s = 0.0f, sq = 0.0f;
    for (int i = threadIdx.x; i < L_; i += 256) { float v = x[i]; s += v; sq += v * v; }
    s1[threadIdx.x] = s; s2[threadIdx.x] = sq;
    __syncthreads();
    for (int o = 128; o > 0; o >>= 1) {
        if (threadIdx.x < o) { s1[threadIdx.x] += s1[threadIdx.x + o]; s2[threadIdx.x] += s2[threadIdx.x + o]; }
        __syncthreads();
    }
    float mean = s1[0] * (1.0f / L_);
    float var  = s2[0] * (1.0f / L_) - mean * mean;
    float inv  = rsqrtf(var + 1.0e-5f);
    half_t* y = Yh + (size_t)bc * L_;
    for (int i = threadIdx.x; i < L_; i += 256) {
        float v = (x[i] - mean) * inv;
        y[i] = (half_t)(v > 0.0f ? v : 0.0f);
    }
}

// Instance norm + residual add: out[b,c,:] = F_c + inorm(X)
__global__ void inorm_res_out_kernel(const float* __restrict__ X,
                                     const float* __restrict__ inp,
                                     float* __restrict__ out) {
    int bc = blockIdx.x;
    int b = bc >> 8, c = bc & (C_ - 1);
    const float* x = X + (size_t)bc * L_;
    __shared__ float s1[256], s2[256];
    float s = 0.0f, sq = 0.0f;
    for (int i = threadIdx.x; i < L_; i += 256) { float v = x[i]; s += v; sq += v * v; }
    s1[threadIdx.x] = s; s2[threadIdx.x] = sq;
    __syncthreads();
    for (int o = 128; o > 0; o >>= 1) {
        if (threadIdx.x < o) { s1[threadIdx.x] += s1[threadIdx.x + o]; s2[threadIdx.x] += s2[threadIdx.x + o]; }
        __syncthreads();
    }
    float mean = s1[0] * (1.0f / L_);
    float var  = s2[0] * (1.0f / L_) - mean * mean;
    float inv  = rsqrtf(var + 1.0e-5f);
    const float* fc = inp + ((size_t)b * C2_ + c) * L_;
    float* o = out + ((size_t)b * C2_ + c) * L_;
    for (int i = threadIdx.x; i < L_; i += 256) o[i] = fc[i] + (x[i] - mean) * inv;
}

// out[:, C:2C] = F_s
__global__ void copy_fs_kernel(const float* __restrict__ inp, float* __restrict__ out) {
    size_t t = (size_t)blockIdx.x * 256 + threadIdx.x;   // over B*C*L
    int l = (int)(t & (L_ - 1));
    size_t bc = t >> 12;
    int c = (int)(bc & (C_ - 1));
    int b = (int)(bc >> 8);
    size_t off = ((size_t)b * C2_ + C_ + c) * L_ + l;
    out[off] = inp[off];
}

// ---------------------------------------------------------------------------
extern "C" void kernel_launch(void* const* d_in, const int* in_sizes, int n_in,
                              void* d_out, int out_size, void* d_ws, size_t ws_size,
                              hipStream_t stream) {
    const float* inp  = (const float*)d_in[0];   // (8,512,64,64) f32
    const int*   mask = (const int*)d_in[1];     // (64,64) i32
    const float* W1   = (const float*)d_in[2];   // (256,512,3,3)
    const float* b1   = (const float*)d_in[3];   // (256)
    const float* W2   = (const float*)d_in[4];   // (256,256,3,3)
    const float* b2   = (const float*)d_in[5];   // (256)
    float* out = (float*)d_out;

    char* ws = (char*)d_ws;
    int*    idx   = (int*)(ws + OFF_IDX);
    half_t* W1h   = (half_t*)(ws + OFF_W1H);
    half_t* W2h   = (half_t*)(ws + OFF_W2H);
    half_t* latn  = (half_t*)(ws + OFF_LATN);   // reused as h (conv2 input) after sim
    half_t* Fcom  = (half_t*)(ws + OFF_FCOM);   // reused as conv2-out (f32) after conv1
    float*  c1out = (float*)(ws + OFF_C1O);
    half_t* hbuf  = latn;
    float*  c2out = (float*)(ws + OFF_FCOM);

    const int nW1 = C_ * C2_ * 9;   // 1,179,648
    const int nW2 = C_ * C_ * 9;    //   589,824

    cvt_f16_kernel<<<(nW1 + 255) / 256, 256, 0, stream>>>(W1, W1h, nW1);
    cvt_f16_kernel<<<(nW2 + 255) / 256, 256, 0, stream>>>(W2, W2h, nW2);

    normalize_kernel<<<(B_ * L_) / 256, 256, 0, stream>>>(inp, latn);

    sim_argmax_kernel<<<B_ * (L_ / 128), 256, 0, stream>>>(latn, mask, idx);

    fuse_kernel<<<(B_ * C_ * L_) / 256, 256, 0, stream>>>(inp, mask, idx, Fcom);

    // conv1: Cin=512, K=4608
    conv_wmma_kernel<<<B_ * 4 * (L_ / 64), 256, 0, stream>>>(Fcom, W1h, b1, c1out, C2_);

    inorm_relu_kernel<<<B_ * C_, 256, 0, stream>>>(c1out, hbuf);

    // conv2: Cin=256, K=2304 (writes into Fcom region; Fcom dead)
    conv_wmma_kernel<<<B_ * 4 * (L_ / 64), 256, 0, stream>>>(hbuf, W2h, b2, c2out, C_);

    inorm_res_out_kernel<<<B_ * C_, 256, 0, stream>>>(c2out, inp, out);

    copy_fs_kernel<<<(B_ * C_ * L_) / 256, 256, 0, stream>>>(inp, out);
}
